// ReccurentGNN_76879914598476
// MI455X (gfx1250) — matrix operand order repaired and checked by reading, not compiled, not run on hardware
//
#include <hip/hip_runtime.h>
#include <hip/hip_bf16.h>
#include <cstdint>
#include <cstddef>

#define EMB 128
#define GNN_RADIUS 0.075f

// ---------------------------------------------------------------------------
// CDNA5 WMMA types
// ---------------------------------------------------------------------------
typedef __attribute__((ext_vector_type(16))) __bf16 v16bf;
typedef __attribute__((ext_vector_type(8)))  float  v8f;

union FragBF {
  v16bf v;
  uint4 q[2];
};

// f32 -> bf16 via native converter (v_cvt_pk_bf16_f32, RNE).
__device__ __forceinline__ unsigned short f2bf(float f) {
  union { __bf16 b; unsigned short u; } cv;
  cv.b = (__bf16)f;
  return cv.u;
}
// Two f32 -> one packed dword (single v_cvt_pk_bf16_f32).
__device__ __forceinline__ unsigned int f2bf_pk(float lo, float hi) {
  return (unsigned int)f2bf(lo) | ((unsigned int)f2bf(hi) << 16);
}
__device__ __forceinline__ float bf2f(unsigned short u) {
  return __uint_as_float(((unsigned int)u) << 16);
}

// ---------------------------------------------------------------------------
// Weight conversion: f32 [rows, cin] -> bf16 [rows, cinPad] (zero padded K)
// ---------------------------------------------------------------------------
__global__ __launch_bounds__(256) void cvt_weight_kernel(
    const float* __restrict__ src, unsigned short* __restrict__ dst,
    int rows, int cin, int cinPad) {
  int i = blockIdx.x * 256 + threadIdx.x;
  int total = rows * cinPad;
  if (i >= total) return;
  int r = i / cinPad, c = i - r * cinPad;
  dst[i] = (c < cin) ? f2bf(src[(size_t)r * cin + c]) : (unsigned short)0;
}

// ---------------------------------------------------------------------------
// Node features: nf[n] = [x0,x1,x2,x3,x4,x4,x4,x4, 0...] padded to 32 (bf16)
// ---------------------------------------------------------------------------
__global__ __launch_bounds__(256) void build_nf_kernel(
    const float* __restrict__ x, unsigned short* __restrict__ nf, int Nn) {
  int n = blockIdx.x * 256 + threadIdx.x;
  if (n >= Nn) return;
  const float* xr = x + (size_t)n * 5;
  unsigned short o[32];
#pragma unroll
  for (int q = 0; q < 32; ++q) o[q] = 0;
  o[0] = f2bf(xr[0]); o[1] = f2bf(xr[1]);
  o[2] = f2bf(xr[2]); o[3] = f2bf(xr[3]);
  unsigned short a4 = f2bf(xr[4]);
  o[4] = a4; o[5] = a4; o[6] = a4; o[7] = a4;
  unsigned short* d = nf + (size_t)n * 32;
#pragma unroll
  for (int q = 0; q < 32; ++q) d[q] = o[q];
}

// ---------------------------------------------------------------------------
// Edge features (11-d, padded to 32, bf16): gather endpoint rows of x
// ---------------------------------------------------------------------------
__global__ __launch_bounds__(256) void build_ef_kernel(
    const float* __restrict__ x, const int* __restrict__ e_src,
    const int* __restrict__ e_dst, const float* __restrict__ vnorm,
    unsigned short* __restrict__ ef, int Ee) {
  int e = blockIdx.x * 256 + threadIdx.x;
  if (e >= Ee) return;
  int i = e_dst[e], j = e_src[e];
  const float* xi = x + (size_t)i * 5;
  const float* xj = x + (size_t)j * 5;
  float dx = xi[0] - xj[0], dy = xi[1] - xj[1];
  const float inv = 1.0f / GNN_RADIUS;
  float r = sqrtf(dx * dx + dy * dy) * inv;
  float v4 = vnorm[4], v5 = vnorm[5];
  unsigned short o[32];
#pragma unroll
  for (int q = 0; q < 32; ++q) o[q] = 0;
  o[0] = f2bf(dx * inv); o[1] = f2bf(dy * inv); o[2] = f2bf(r);
  o[3] = f2bf(xi[2] / v4); o[4] = f2bf(xi[3] / v5);
  o[5] = f2bf(xj[2] / v4); o[6] = f2bf(xj[3] / v5);
  unsigned short a4 = f2bf(xi[4]);
  o[7] = a4; o[8] = a4; o[9] = a4; o[10] = a4;
  unsigned short* d = ef + (size_t)e * 32;
#pragma unroll
  for (int q = 0; q < 32; ++q) d[q] = o[q];
}

// ---------------------------------------------------------------------------
// WMMA tile: acc(16x16 f32) = A_lds(16x K) @ W^T(K x 16), K multiple of 32.
//   A layout (16-bit A 16x32 per step): lane<16 holds K {k0..k0+7, k0+16..23},
//   lane>=16 holds K {k0+8..15, k0+24..31} for row M = lane&15.
//   B: lane&15 = column n -> weight row (nt*16+n); same K split as A.
// ---------------------------------------------------------------------------
template <int K>
__device__ __forceinline__ v8f compute_tile(
    const unsigned short* A,                     // LDS, 32 x K bf16
    const unsigned short* __restrict__ W,        // global, [NOUT, K] bf16
    int mt, int nt, int lane) {
  const int hl  = lane >> 4;
  const int l16 = lane & 15;
  const unsigned short* arow = A + (size_t)(mt * 16 + l16) * K + hl * 8;
  const unsigned short* wrow = W + (size_t)(nt * 16 + l16) * K + hl * 8;
  v8f acc = {};
#pragma unroll
  for (int k0 = 0; k0 < K; k0 += 32) {
    FragBF a, b;
    a.q[0] = *(const uint4*)(arow + k0);
    a.q[1] = *(const uint4*)(arow + k0 + 16);
    b.q[0] = *(const uint4*)(wrow + k0);
    b.q[1] = *(const uint4*)(wrow + k0 + 16);
    acc = __builtin_amdgcn_wmma_f32_16x16x32_bf16(
        false, a.v, false, b.v, (short)0, acc, false, false);
  }
  return acc;
}

// Hidden writeback: bias + ReLU + paired bf16 pack (1 cvt_pk per 2 rows),
// stores from lo/hi halves (ds_store_b16 / ds_store_b16_d16_hi).
template <int NOUT>
__device__ __forceinline__ void store_hidden(
    unsigned short* H, v8f acc, const float* __restrict__ bias,
    int mt, int nt, int lane) {
  const int hl = lane >> 4;
  const int n  = lane & 15;
  const int col = nt * 16 + n;
  const float bv = bias[col];
  unsigned short* base = H + (size_t)(mt * 16 + hl * 8) * NOUT + col;
#pragma unroll
  for (int r = 0; r < 8; r += 2) {
    float v0 = fmaxf(acc[r] + bv, 0.0f);
    float v1 = fmaxf(acc[r + 1] + bv, 0.0f);
    unsigned int pk = f2bf_pk(v0, v1);
    base[(size_t)r * NOUT]       = (unsigned short)pk;          // lo half
    base[(size_t)(r + 1) * NOUT] = (unsigned short)(pk >> 16);  // d16_hi
  }
}

// ---------------------------------------------------------------------------
// Fused 3-layer MLP (Linear-ReLU-Linear-ReLU-Linear), 32 rows per block,
// 8 waves: wave w -> M-tile (w&1), N-tiles (w>>1) stride 4. Hidden layers
// never leave LDS. Output width is always EMB=128.
//   INM: 0 plain bf16 [M,K0]; 1 edge gather cat(edge, node[dst], node[src]);
//        2 node cat(node_bf16, bf16(aggr_f32))
//   OUTM: 0 store f32(optional)+bf16; 1 edge residual + atomic segment-sum;
//         2 node residual (f32 master) + bf16 shadow
// ---------------------------------------------------------------------------
enum { IN_PLAIN = 0, IN_EDGE = 1, IN_NODE = 2 };
enum { OUT_PLAIN = 0, OUT_EDGE = 1, OUT_NODE = 2 };

template <int K0, int H, int INM, int OUTM>
__global__ __launch_bounds__(256) void fused_mlp3_kernel(
    const unsigned short* __restrict__ W0, const float* __restrict__ b0,
    const unsigned short* __restrict__ W1, const float* __restrict__ b1,
    const unsigned short* __restrict__ W2, const float* __restrict__ b2,
    const unsigned short* A_plain,
    const unsigned short* edge_bf, unsigned short* edge_bf_out,
    const unsigned short* node_bf_in,
    const int* __restrict__ e_src, const int* __restrict__ e_dst,
    const float* __restrict__ aggr_in, float* aggr_out,
    float* node_f32_io, float* out_f32, unsigned short* out_bf16,
    int M) {
  constexpr int BUF0 = (K0 > H ? K0 : H);
  __shared__ alignas(16) unsigned short sb0[32 * BUF0];
  __shared__ alignas(16) unsigned short sb1[32 * H];
  __shared__ int sDst[32];

  const int tid  = threadIdx.x;
  const int lane = tid & 31;
  const int wave = tid >> 5;
  const int mt   = wave & 1;
  const int row0 = blockIdx.x * 32;

  // ---- stage 0: build A tile (32 x K0) in LDS, bf16, zero-pad tail rows ----
  if constexpr (INM == IN_PLAIN) {
    constexpr int SEGS = K0 / 8;
    for (int c = tid; c < 32 * SEGS; c += 256) {
      int row = c / SEGS, seg = c - row * SEGS;
      int g = row0 + row;
      uint4 v = make_uint4(0, 0, 0, 0);
      if (g < M) v = *(const uint4*)(A_plain + (size_t)g * K0 + seg * 8);
      *(uint4*)(sb0 + (size_t)row * K0 + seg * 8) = v;
    }
  } else if constexpr (INM == IN_EDGE) {
    if (tid < 32) {
      int e = row0 + tid;
      sDst[tid] = (e < M) ? e_dst[e] : 0;
    }
    constexpr int SEGS = K0 / 8;  // 48: [edge | node[dst] | node[src]]
    for (int c = tid; c < 32 * SEGS; c += 256) {
      int row = c / SEGS, seg = c - row * SEGS;
      int e = row0 + row;
      uint4 v = make_uint4(0, 0, 0, 0);
      if (e < M) {
        if (seg < 16)
          v = *(const uint4*)(edge_bf + (size_t)e * EMB + seg * 8);
        else if (seg < 32)
          v = *(const uint4*)(node_bf_in + (size_t)e_dst[e] * EMB + (seg - 16) * 8);
        else
          v = *(const uint4*)(node_bf_in + (size_t)e_src[e] * EMB + (seg - 32) * 8);
      }
      *(uint4*)(sb0 + (size_t)row * K0 + seg * 8) = v;
    }
  } else {  // IN_NODE : [node_bf16 | bf16(aggr_f32)]
    constexpr int SEGS = K0 / 8;  // 32
    for (int c = tid; c < 32 * SEGS; c += 256) {
      int row = c / SEGS, seg = c - row * SEGS;
      int g = row0 + row;
      uint4 v = make_uint4(0, 0, 0, 0);
      if (g < M) {
        if (seg < 16) {
          v = *(const uint4*)(node_bf_in + (size_t)g * EMB + seg * 8);
        } else {
          const float* s = aggr_in + (size_t)g * EMB + (seg - 16) * 8;
          v.x = f2bf_pk(s[0], s[1]);
          v.y = f2bf_pk(s[2], s[3]);
          v.z = f2bf_pk(s[4], s[5]);
          v.w = f2bf_pk(s[6], s[7]);
        }
      }
      *(uint4*)(sb0 + (size_t)row * K0 + seg * 8) = v;
    }
  }
  __syncthreads();

  // ---- GEMM1: (32 x K0) @ (K0 x H) -> sb1 (ReLU) ----
  for (int nt = wave >> 1; nt < H / 16; nt += 4) {
    __builtin_prefetch(W1 + (size_t)(nt * 16) * H, 0, 1);
    v8f acc = compute_tile<K0>(sb0, W0, mt, nt, lane);
    store_hidden<H>(sb1, acc, b0, mt, nt, lane);
  }
  __syncthreads();

  // ---- GEMM2: (32 x H) @ (H x H) -> sb0 (ReLU) ----
  for (int nt = wave >> 1; nt < H / 16; nt += 4) {
    __builtin_prefetch(W2 + (size_t)(nt * 16) * H, 0, 1);
    v8f acc = compute_tile<H>(sb1, W1, mt, nt, lane);
    store_hidden<H>(sb0, acc, b1, mt, nt, lane);
  }
  __syncthreads();

  // ---- GEMM3: (32 x H) @ (H x 128) -> epilogue (no ReLU) ----
  for (int nt = wave >> 1; nt < EMB / 16; nt += 4) {
    v8f acc = compute_tile<H>(sb0, W2, mt, nt, lane);
    const int hl  = lane >> 4;
    const int n   = lane & 15;
    const int col = nt * 16 + n;
    const float bv = b2[col];
#pragma unroll
    for (int r = 0; r < 8; r += 2) {
      const int row = mt * 16 + hl * 8 + r;
      const int g = row0 + row;
      float v0 = acc[r] + bv;
      float v1 = acc[r + 1] + bv;
      if constexpr (OUTM == OUT_PLAIN) {
        if (g < M) {
          if (out_f32) out_f32[(size_t)g * EMB + col] = v0;
          out_bf16[(size_t)g * EMB + col] = f2bf(v0);
        }
        if (g + 1 < M) {
          if (out_f32) out_f32[(size_t)(g + 1) * EMB + col] = v1;
          out_bf16[(size_t)(g + 1) * EMB + col] = f2bf(v1);
        }
      } else if constexpr (OUTM == OUT_EDGE) {
        // edge += msg ; aggr[dst] += msg (segment-sum via HW f32 atomics)
        if (g < M) {
          float enew = bf2f(edge_bf[(size_t)g * EMB + col]) + v0;
          edge_bf_out[(size_t)g * EMB + col] = f2bf(enew);
          unsafeAtomicAdd(&aggr_out[(size_t)sDst[row] * EMB + col], v0);
        }
        if (g + 1 < M) {
          float enew = bf2f(edge_bf[(size_t)(g + 1) * EMB + col]) + v1;
          edge_bf_out[(size_t)(g + 1) * EMB + col] = f2bf(enew);
          unsafeAtomicAdd(&aggr_out[(size_t)sDst[row + 1] * EMB + col], v1);
        }
      } else {  // OUT_NODE: node += mlp(...), fp32 master + bf16 shadow
        if (g < M) {
          float nv = node_f32_io[(size_t)g * EMB + col] + v0;
          node_f32_io[(size_t)g * EMB + col] = nv;
          out_bf16[(size_t)g * EMB + col] = f2bf(nv);
        }
        if (g + 1 < M) {
          float nv = node_f32_io[(size_t)(g + 1) * EMB + col] + v1;
          node_f32_io[(size_t)(g + 1) * EMB + col] = nv;
          out_bf16[(size_t)(g + 1) * EMB + col] = f2bf(nv);
        }
      }
    }
  }
}

// ---------------------------------------------------------------------------
// Output head: 128 -> 32 -> 32 -> 2, fp32 VALU (negligible FLOPs)
// ---------------------------------------------------------------------------
__global__ __launch_bounds__(128) void node_out_kernel(
    const float* __restrict__ h,
    const float* __restrict__ W0, const float* __restrict__ b0,
    const float* __restrict__ W1, const float* __restrict__ b1,
    const float* __restrict__ W2, const float* __restrict__ b2,
    float* __restrict__ out, int Nn) {
  int n = blockIdx.x * 128 + threadIdx.x;
  if (n >= Nn) return;
  const float* hr = h + (size_t)n * EMB;
  float h1[32];
#pragma unroll
  for (int o = 0; o < 32; ++o) h1[o] = b0[o];
  for (int k = 0; k < EMB; ++k) {
    float xv = hr[k];
#pragma unroll
    for (int o = 0; o < 32; ++o) h1[o] = fmaf(W0[o * EMB + k], xv, h1[o]);
  }
  float h2[32];
#pragma unroll
  for (int o = 0; o < 32; ++o) h2[o] = b1[o];
#pragma unroll
  for (int k = 0; k < 32; ++k) {
    float xv = fmaxf(h1[k], 0.0f);
#pragma unroll
    for (int o = 0; o < 32; ++o) h2[o] = fmaf(W1[o * 32 + k], xv, h2[o]);
  }
  float o0 = b2[0], o1 = b2[1];
#pragma unroll
  for (int k = 0; k < 32; ++k) {
    float xv = fmaxf(h2[k], 0.0f);
    o0 = fmaf(W2[k], xv, o0);
    o1 = fmaf(W2[32 + k], xv, o1);
  }
  out[(size_t)n * 2 + 0] = o0;
  out[(size_t)n * 2 + 1] = o1;
}

// ---------------------------------------------------------------------------
// Host driver
// ---------------------------------------------------------------------------
extern "C" void kernel_launch(void* const* d_in, const int* in_sizes, int n_in,
                              void* d_out, int out_size, void* d_ws, size_t ws_size,
                              hipStream_t stream) {
  (void)n_in; (void)out_size; (void)ws_size;

  const float* x     = (const float*)d_in[0];
  const int*   eidx  = (const int*)d_in[1];
  const float* vnorm = (const float*)d_in[3];
  const int N = in_sizes[0] / 5;
  const int E = in_sizes[1] / 2;
  const int* e_src = eidx;        // edge_index[0]
  const int* e_dst = eidx + E;    // edge_index[1]

  const float *en_W0 = (const float*)d_in[4],  *en_b0 = (const float*)d_in[5];
  const float *en_W1 = (const float*)d_in[6],  *en_b1 = (const float*)d_in[7];
  const float *en_W2 = (const float*)d_in[8],  *en_b2 = (const float*)d_in[9];
  const float *ee_W0 = (const float*)d_in[10], *ee_b0 = (const float*)d_in[11];
  const float *ee_W1 = (const float*)d_in[12], *ee_b1 = (const float*)d_in[13];
  const float *ee_W2 = (const float*)d_in[14], *ee_b2 = (const float*)d_in[15];
  const float *le_W0 = (const float*)d_in[16], *le_b0 = (const float*)d_in[17];
  const float *le_W1 = (const float*)d_in[18], *le_b1 = (const float*)d_in[19];
  const float *le_W2 = (const float*)d_in[20], *le_b2 = (const float*)d_in[21];
  const float *ln_W0 = (const float*)d_in[22], *ln_b0 = (const float*)d_in[23];
  const float *ln_W1 = (const float*)d_in[24], *ln_b1 = (const float*)d_in[25];
  const float *ln_W2 = (const float*)d_in[26], *ln_b2 = (const float*)d_in[27];
  const float *no_W0 = (const float*)d_in[28], *no_b0 = (const float*)d_in[29];
  const float *no_W1 = (const float*)d_in[30], *no_b1 = (const float*)d_in[31];
  const float *no_W2 = (const float*)d_in[32], *no_b2 = (const float*)d_in[33];

  // ---- carve workspace ----
  char* base = (char*)d_ws;
  size_t off = 0;
  auto take = [&](size_t bytes) -> void* {
    void* p = base + off;
    off = (off + bytes + 255) & ~(size_t)255;
    return p;
  };
  unsigned short* node_bf  = (unsigned short*)take((size_t)N * EMB * 2);
  float*          node_f32 = (float*)take((size_t)N * EMB * 4);
  unsigned short* edge_bf  = (unsigned short*)take((size_t)E * EMB * 2);
  float*          aggr     = (float*)take((size_t)N * EMB * 4);
  unsigned short* nf       = (unsigned short*)take((size_t)N * 32 * 2);
  unsigned short* ef       = (unsigned short*)take((size_t)E * 32 * 2);
  unsigned short* w_en0 = (unsigned short*)take((size_t)128 * 32 * 2);
  unsigned short* w_en1 = (unsigned short*)take((size_t)128 * 128 * 2);
  unsigned short* w_en2 = (unsigned short*)take((size_t)128 * 128 * 2);
  unsigned short* w_ee0 = (unsigned short*)take((size_t)128 * 32 * 2);
  unsigned short* w_ee1 = (unsigned short*)take((size_t)128 * 128 * 2);
  unsigned short* w_ee2 = (unsigned short*)take((size_t)128 * 128 * 2);
  unsigned short* w_le0 = (unsigned short*)take((size_t)4 * 384 * 384 * 2);
  unsigned short* w_le1 = (unsigned short*)take((size_t)4 * 384 * 384 * 2);
  unsigned short* w_le2 = (unsigned short*)take((size_t)4 * 128 * 384 * 2);
  unsigned short* w_ln0 = (unsigned short*)take((size_t)4 * 256 * 256 * 2);
  unsigned short* w_ln1 = (unsigned short*)take((size_t)4 * 256 * 256 * 2);
  unsigned short* w_ln2 = (unsigned short*)take((size_t)4 * 128 * 256 * 2);

  auto cvt = [&](const float* src, unsigned short* dst, int rows, int cin, int cinPad) {
    int total = rows * cinPad;
    cvt_weight_kernel<<<(total + 255) / 256, 256, 0, stream>>>(src, dst, rows, cin, cinPad);
  };
  cvt(en_W0, w_en0, 128, 8, 32);
  cvt(en_W1, w_en1, 128, 128, 128);
  cvt(en_W2, w_en2, 128, 128, 128);
  cvt(ee_W0, w_ee0, 128, 11, 32);
  cvt(ee_W1, w_ee1, 128, 128, 128);
  cvt(ee_W2, w_ee2, 128, 128, 128);
  cvt(le_W0, w_le0, 4 * 384, 384, 384);
  cvt(le_W1, w_le1, 4 * 384, 384, 384);
  cvt(le_W2, w_le2, 4 * 128, 384, 384);
  cvt(ln_W0, w_ln0, 4 * 256, 256, 256);
  cvt(ln_W1, w_ln1, 4 * 256, 256, 256);
  cvt(ln_W2, w_ln2, 4 * 128, 256, 256);

  // ---- features ----
  build_nf_kernel<<<(N + 255) / 256, 256, 0, stream>>>(x, nf, N);
  build_ef_kernel<<<(E + 255) / 256, 256, 0, stream>>>(x, e_src, e_dst, vnorm, ef, E);

  // ---- embeddings (WMMA, K padded 8/11 -> 32) ----
  fused_mlp3_kernel<32, 128, IN_PLAIN, OUT_PLAIN><<<(N + 31) / 32, 256, 0, stream>>>(
      w_en0, en_b0, w_en1, en_b1, w_en2, en_b2,
      nf, nullptr, nullptr, nullptr, nullptr, nullptr,
      nullptr, nullptr, nullptr, node_f32, node_bf, N);
  fused_mlp3_kernel<32, 128, IN_PLAIN, OUT_PLAIN><<<(E + 31) / 32, 256, 0, stream>>>(
      w_ee0, ee_b0, w_ee1, ee_b1, w_ee2, ee_b2,
      ef, nullptr, nullptr, nullptr, nullptr, nullptr,
      nullptr, nullptr, nullptr, nullptr, edge_bf, E);

  // ---- message-passing layers ----
  for (int l = 0; l < 4; ++l) {
    hipMemsetAsync(aggr, 0, (size_t)N * EMB * 4, stream);
    fused_mlp3_kernel<384, 384, IN_EDGE, OUT_EDGE><<<(E + 31) / 32, 256, 0, stream>>>(
        w_le0 + (size_t)l * 384 * 384, le_b0 + (size_t)l * 384,
        w_le1 + (size_t)l * 384 * 384, le_b1 + (size_t)l * 384,
        w_le2 + (size_t)l * 128 * 384, le_b2 + (size_t)l * 128,
        nullptr, edge_bf, edge_bf, node_bf, e_src, e_dst,
        nullptr, aggr, nullptr, nullptr, nullptr, E);
    fused_mlp3_kernel<256, 256, IN_NODE, OUT_NODE><<<(N + 31) / 32, 256, 0, stream>>>(
        w_ln0 + (size_t)l * 256 * 256, ln_b0 + (size_t)l * 256,
        w_ln1 + (size_t)l * 256 * 256, ln_b1 + (size_t)l * 256,
        w_ln2 + (size_t)l * 128 * 256, ln_b2 + (size_t)l * 128,
        nullptr, nullptr, nullptr, node_bf, nullptr, nullptr,
        aggr, nullptr, node_f32, nullptr, node_bf, N);
  }

  // ---- output head ----
  node_out_kernel<<<(N + 127) / 128, 128, 0, stream>>>(
      node_f32, no_W0, no_b0, no_W1, no_b1, no_W2, no_b2, (float*)d_out, N);
}
